// CrossAttentionLayer_54855322304785
// MI455X (gfx1250) — matrix operand order repaired and checked by reading, not compiled, not run on hardware
//
#include <hip/hip_runtime.h>
#include <stdint.h>

// ---------------------------------------------------------------------------
// CrossAttentionLayer for MI455X (gfx1250, wave32, WMMA).
// B=4, SQ=SK=2048, D=512, H=8, HD=64.  Outputs: out [B,SQ,D] fp32 then
// attn [B,H,SQ,SK] fp32, concatenated in d_out.
//
// bf16 WMMA datapath (fp32 accumulate). Fused scores+softmax+attn@V so the
// 536MB attn tensor is written to HBM once and consumed from LDS. TDM
// tensor_load_to_lds for the Q tile. Projection GEMM stages the W panel in
// LDS as bf16 (once per block). All operand streams use ping-pong double
// buffering with the final pair peeled: no rotation movs, no select overhead.
// Workspace: 64 MB (qlin f32 | klin bf16 | vT bf16 | ctx f32 | oproj f32).
// ---------------------------------------------------------------------------

#define Bn   4
#define SQn  2048
#define SKn  2048
#define Dn   512
#define Hn   8
#define HDn  64
#define SCALEF 0.125f   // 64^-0.5
#define SSTR 2052       // padded LDS row stride (floats) for score strip
#define WSTR 528        // padded LDS row stride (bf16) for W panel

typedef __bf16 bf16x16 __attribute__((ext_vector_type(16)));
typedef __bf16 bf16x8  __attribute__((ext_vector_type(8)));
typedef float  f32x8   __attribute__((ext_vector_type(8)));
typedef float  f32x4   __attribute__((ext_vector_type(4)));
typedef unsigned int v4u __attribute__((ext_vector_type(4)));
typedef int v8i __attribute__((ext_vector_type(8)));
typedef int v4i __attribute__((ext_vector_type(4)));

__device__ __forceinline__ f32x8 wmma_bf16(bf16x16 a, bf16x16 b, f32x8 c) {
  return __builtin_amdgcn_wmma_f32_16x16x32_bf16(false, a, false, b,
                                                 (short)0, c, false, false);
}

__device__ __forceinline__ f32x8 fzero8() {
  f32x8 z;
#pragma unroll
  for (int i = 0; i < 8; ++i) z[i] = 0.f;
  return z;
}

// Load one 32-wide fp32 A chunk (per-lane 16 floats in WMMA A order).
__device__ __forceinline__ void load_chunk(f32x4 (&d)[4], const float* Arow,
                                           int k0, int g) {
  const f32x4* p0 = (const f32x4*)(Arow + k0 + g * 8);
  const f32x4* p1 = (const f32x4*)(Arow + k0 + 16 + g * 8);
  d[0] = p0[0];
  d[1] = p0[1];
  d[2] = p1[0];
  d[3] = p1[1];
}

// One k-step of the projection GEMM: convert A chunk, gather 4 B fragments
// from the LDS W panel, 4 independent WMMAs.
__device__ __forceinline__ void gemm_step(const f32x4 (&ar)[4],
                                          const __bf16* lw, int k0, int lm,
                                          int g, f32x8 (&acc)[4]) {
  bf16x16 a;
#pragma unroll
  for (int i = 0; i < 4; ++i) {
    a[i]      = (__bf16)ar[0][i];
    a[4 + i]  = (__bf16)ar[1][i];
    a[8 + i]  = (__bf16)ar[2][i];
    a[12 + i] = (__bf16)ar[3][i];
  }
  bf16x16 bfr[4];
#pragma unroll
  for (int j = 0; j < 4; ++j) {
    const bf16x8* b8 =
        (const bf16x8*)(lw + (j * 16 + lm) * WSTR + k0 + g * 8);
    const bf16x8 lo = b8[0], hi = b8[2];
#pragma unroll
    for (int i = 0; i < 8; ++i) {
      bfr[j][i]     = lo[i];
      bfr[j][8 + i] = hi[i];
    }
  }
#pragma unroll
  for (int j = 0; j < 4; ++j) acc[j] = wmma_bf16(a, bfr[j], acc[j]);
}

// ---------------------------------------------------------------------------
// Projection GEMM:  C[M=8192, 512] = A[8192,512] @ W[512,512]^T + bias.
// Block = 256 thr (8 waves), tile 128(M) x 64(N); wave: 16(M) x 64(N).
// Optional outputs: Cf fp32 row-major, Cb bf16 row-major,
// CbT bf16 head-transposed [B,H,HD,S] (for V).
// ---------------------------------------------------------------------------
__global__ __launch_bounds__(256, 1) void proj_gemm_kernel(
    const float* __restrict__ A, const float* __restrict__ W,
    const float* __restrict__ bias, float* __restrict__ Cf,
    __bf16* __restrict__ Cb, __bf16* __restrict__ CbT) {
  extern __shared__ char smem[];
  __bf16* lw = (__bf16*)smem;  // 64 x WSTR bf16 = 67,584 B

  const int tid = threadIdx.x;
  const int wv = tid >> 5, l = tid & 31, lm = l & 15, g = l >> 4;
  const int mbase = blockIdx.y * 128 + wv * 16;
  const int nbase = blockIdx.x * 64;

  // ---- Stage W panel rows [nbase, nbase+64) x K into LDS as bf16 ----------
  {
    const f32x4* wp = (const f32x4*)(W + (size_t)nbase * Dn);
#pragma unroll
    for (int it = 0; it < 32; ++it) {
      const int i4 = tid + it * 256;       // f32x4 index in 64x512 panel
      const f32x4 wd = wp[i4];
      const int e = i4 * 4;
      const int row = e >> 9, col = e & 511;
      __bf16* dst = lw + row * WSTR + col;
      dst[0] = (__bf16)wd[0];
      dst[1] = (__bf16)wd[1];
      dst[2] = (__bf16)wd[2];
      dst[3] = (__bf16)wd[3];
    }
  }
  __syncthreads();

  f32x8 acc[4];
#pragma unroll
  for (int j = 0; j < 4; ++j) acc[j] = fzero8();

  const float* Arow = A + (size_t)(mbase + lm) * Dn;

  // Ping-pong double-buffered A stream; final pair peeled.
  f32x4 bufA[4], bufB[4];
  load_chunk(bufA, Arow, 0, g);
  for (int k0 = 0; k0 < Dn - 64; k0 += 64) {
    load_chunk(bufB, Arow, k0 + 32, g);
    if (k0 + 160 < Dn) __builtin_prefetch(Arow + k0 + 160, 0, 1);
    gemm_step(bufA, lw, k0, lm, g, acc);
    load_chunk(bufA, Arow, k0 + 64, g);
    gemm_step(bufB, lw, k0 + 32, lm, g, acc);
  }
  load_chunk(bufB, Arow, Dn - 32, g);
  gemm_step(bufA, lw, Dn - 64, lm, g, acc);
  gemm_step(bufB, lw, Dn - 32, lm, g, acc);

  // ---- Epilogue: uniform output-mode checks hoisted out of loops ----------
  if (Cf) {
#pragma unroll
    for (int j = 0; j < 4; ++j) {
      const int col = nbase + j * 16 + lm;
      const float bsv = bias[col];
#pragma unroll
      for (int r = 0; r < 8; ++r)
        Cf[(size_t)(mbase + g * 8 + r) * Dn + col] = acc[j][r] + bsv;
    }
  }
  if (Cb) {
#pragma unroll
    for (int j = 0; j < 4; ++j) {
      const int col = nbase + j * 16 + lm;
      const float bsv = bias[col];
#pragma unroll
      for (int r = 0; r < 8; ++r)
        Cb[(size_t)(mbase + g * 8 + r) * Dn + col] =
            (__bf16)(acc[j][r] + bsv);
    }
  }
  if (CbT) {
#pragma unroll
    for (int j = 0; j < 4; ++j) {
      const int col = nbase + j * 16 + lm;
      const float bsv = bias[col];
      const int h_ = col >> 6, d_ = col & 63;
#pragma unroll
      for (int r = 0; r < 8; ++r) {
        const int row = mbase + g * 8 + r;
        const int b_ = row >> 11, s_ = row & 2047;
        CbT[(((size_t)(b_ * Hn + h_) * HDn + d_) * (size_t)SKn) + s_] =
            (__bf16)(acc[j][r] + bsv);
      }
    }
  }
}

// Load one 16-col K tile as two B fragments (K-dim chunks 0..31 / 32..63).
__device__ __forceinline__ void load_kfrag(bf16x16& o0, bf16x16& o1,
                                           const __bf16* kbase, int col,
                                           int lm, int g) {
  const __bf16* kp = kbase + (size_t)(col + lm) * Dn + g * 16;
  o0 = *(const bf16x16*)kp;
  o1 = *(const bf16x16*)(kp + 32);
}

// One 16-col score tile: 2 WMMAs over HD=64, scaled store into LDS strip.
__device__ __forceinline__ void score_tile(const bf16x16& a0, const bf16x16& a1,
                                           const bf16x16& b0, const bf16x16& b1,
                                           float* sc, int cb, int lm, int g) {
  f32x8 s = fzero8();
  s = wmma_bf16(a0, b0, s);
  s = wmma_bf16(a1, b1, s);
#pragma unroll
  for (int r = 0; r < 8; ++r)
    sc[(g * 8 + r) * SSTR + cb + lm] = s[r] * SCALEF;
}

// Load the 4 V fragments (N-subtiles of HD=64) for one 32-wide k chunk.
__device__ __forceinline__ void load_vfrag(bf16x16 (&d)[4],
                                           const __bf16* vbase, int kk, int lm,
                                           int g) {
#pragma unroll
  for (int j = 0; j < 4; ++j)
    d[j] =
        *(const bf16x16*)(vbase + (size_t)(j * 16 + lm) * SKn + kk + g * 16);
}

// One 32-k step of attn@V: A fragment from LDS attn (fp32->bf16), 4 WMMAs.
__device__ __forceinline__ void av_step(const float* sc, int kk, int lm, int g,
                                        const bf16x16 (&vb)[4],
                                        f32x8 (&o)[4]) {
  bf16x16 af;
  const float* sr = sc + lm * SSTR + kk + g * 16;
#pragma unroll
  for (int i = 0; i < 8; ++i) {
    af[i]     = (__bf16)sr[i];
    af[8 + i] = (__bf16)sr[8 + i];
  }
#pragma unroll
  for (int j = 0; j < 4; ++j) o[j] = wmma_bf16(af, vb[j], o[j]);
}

// ---------------------------------------------------------------------------
// Fused attention: per block (b, h, 16 q-rows):
//   TDM-load Q tile (fp32) -> LDS, convert bf16
//   scores (WMMA bf16, ping-pong K loads) -> LDS fp32 strip [16 x 2048]
//   softmax in LDS, write attn to d_out (only HBM touch of attn)
//   attn @ V (WMMA bf16, attn from LDS, ping-pong V loads) -> ctx fp32
// Block 256 thr (8 waves). Dynamic LDS = 170,496 B.
// ---------------------------------------------------------------------------
__global__ __launch_bounds__(256, 1) void attn_kernel(
    const float* __restrict__ qlin, const __bf16* __restrict__ klin,
    const __bf16* __restrict__ vT, float* __restrict__ ctx,
    float* __restrict__ attn_out) {
  extern __shared__ char smem[];
  float*  qf   = (float*)smem;                         // 16x64 fp32 (4 KB)
  __bf16* qb   = (__bf16*)(smem + 4096);               // 16x72 bf16 (2.25 KB)
  float*  sc   = (float*)(smem + 6400);                // 16 x SSTR fp32 (128 KB)
  float*  part = (float*)(smem + 6400 + 16 * SSTR * 4);// 8 x 16 x 64 fp32 (32 KB)

  const int tid = threadIdx.x;
  const int wv = tid >> 5, l = tid & 31, lm = l & 15, g = l >> 4;
  const int bb = blockIdx.z, hh = blockIdx.y, q0 = blockIdx.x * 16;

  const float* qsrc = qlin + ((size_t)(bb * SQn + q0)) * Dn + hh * HDn;

  // ---- Phase 0: Q tile into LDS via Tensor Data Mover ---------------------
#if __has_builtin(__builtin_amdgcn_tensor_load_to_lds) && \
    __has_builtin(__builtin_amdgcn_s_wait_tensorcnt)
  if (wv == 0) {
    // Tensor DMA descriptor (CDNA5 ISA): 2-D fp32 tile,
    // tile 64 x 16, tensor row stride 512 elements.
    const unsigned long long ga = (unsigned long long)(uintptr_t)qsrc;
    const unsigned int ldsoff = (unsigned int)(uintptr_t)qf;
    v4u g0;
    g0[0] = 1u;                                       // count=1, user mode
    g0[1] = ldsoff;                                   // lds_addr (bytes)
    g0[2] = (unsigned int)ga;                         // global_addr[31:0]
    g0[3] = (unsigned int)((ga >> 32) & 0x01FFFFFFu)  // global_addr[56:32]
            | 0x80000000u;                            // type=2 ("image")
    v8i g1;
    g1[0] = 0x00020000;        // wg_mask=0, data_size=2 (4B)
    g1[1] = (64 << 16);        // tensor_dim0 = 64 (lo16)
    g1[2] = (16 << 16);        // tensor_dim0 hi16=0 | tensor_dim1=16 (lo16)
    g1[3] = (64 << 16);        // tensor_dim1 hi16=0 | tile_dim0 = 64
    g1[4] = 16;                // tile_dim1 = 16, tile_dim2 = 0
    g1[5] = 512;               // tensor_dim0_stride lo32 = 512 elements
    g1[6] = 0;
    g1[7] = 0;
    v4i gz4;
    gz4[0] = gz4[1] = gz4[2] = gz4[3] = 0;
    v8i gz8;
#pragma unroll
    for (int i = 0; i < 8; ++i) gz8[i] = 0;
    __builtin_amdgcn_tensor_load_to_lds(g0, g1, gz4, gz4, gz8, 0);
    __builtin_amdgcn_s_wait_tensorcnt(0);
  }
#else
  for (int i = tid; i < 1024; i += 256)
    qf[i] = qsrc[(i >> 6) * Dn + (i & 63)];
#endif
  __syncthreads();
  for (int i = tid; i < 1024; i += 256)
    qb[(i >> 6) * 72 + (i & 63)] = (__bf16)qf[i];
  __syncthreads();

  // ---- Phase 1: scores = (Q K^T) * scale into LDS -------------------------
  bf16x16 a0, a1;  // A fragments for K-dim chunks [0,32) and [32,64)
#pragma unroll
  for (int i = 0; i < 8; ++i) {
    a0[i]     = qb[lm * 72 + g * 16 + i];
    a0[8 + i] = qb[lm * 72 + g * 16 + 8 + i];
    a1[i]     = qb[lm * 72 + 32 + g * 16 + i];
    a1[8 + i] = qb[lm * 72 + 32 + g * 16 + 8 + i];
  }
  const __bf16* kbase = klin + (size_t)(bb * SKn) * Dn + hh * HDn;

  // Ping-pong double-buffered K fragments; final pair peeled.
  bf16x16 b0, b1, c0, c1;
  load_kfrag(b0, b1, kbase, wv * 256, lm, g);
  for (int i = 0; i < 7; ++i) {
    const int cb = wv * 256 + i * 32;
    load_kfrag(c0, c1, kbase, cb + 16, lm, g);
    score_tile(a0, a1, b0, b1, sc, cb, lm, g);
    load_kfrag(b0, b1, kbase, cb + 32, lm, g);
    score_tile(a0, a1, c0, c1, sc, cb + 16, lm, g);
  }
  load_kfrag(c0, c1, kbase, wv * 256 + 240, lm, g);
  score_tile(a0, a1, b0, b1, sc, wv * 256 + 224, lm, g);
  score_tile(a0, a1, c0, c1, sc, wv * 256 + 240, lm, g);
  __syncthreads();

  // ---- Phase 2: softmax rows in LDS, write attn to d_out ------------------
  for (int ro = wv * 2; ro < wv * 2 + 2; ++ro) {
    float mx = -3.0e38f;
    for (int c = l; c < SKn; c += 32) mx = fmaxf(mx, sc[ro * SSTR + c]);
#pragma unroll
    for (int off = 16; off >= 1; off >>= 1)
      mx = fmaxf(mx, __shfl_xor(mx, off, 32));
    float sum = 0.f;
    for (int c = l; c < SKn; c += 32) {
      float e = __expf(sc[ro * SSTR + c] - mx);
      sc[ro * SSTR + c] = e;
      sum += e;
    }
#pragma unroll
    for (int off = 16; off >= 1; off >>= 1) sum += __shfl_xor(sum, off, 32);
    const float inv = 1.f / sum;
    float* arow =
        attn_out + (((size_t)(bb * Hn + hh) * SQn + q0 + ro) * (size_t)SKn);
    for (int c = l; c < SKn; c += 32) {
      float av = sc[ro * SSTR + c] * inv;
      sc[ro * SSTR + c] = av;
      arow[c] = av;
    }
  }
  __syncthreads();

  // ---- Phase 3: ctx = attn @ V (attn from LDS, V bf16 head-transposed) ----
  f32x8 o[4];
#pragma unroll
  for (int j = 0; j < 4; ++j) o[j] = fzero8();

  const __bf16* vbase = vT + (size_t)((bb * Hn + hh) * HDn) * (size_t)SKn;

  // Ping-pong double-buffered V fragments; final pair peeled.
  bf16x16 u[4], w2[4];
  load_vfrag(u, vbase, wv * 256, lm, g);
  for (int i = 0; i < 3; ++i) {
    const int kk = wv * 256 + i * 64;
    load_vfrag(w2, vbase, kk + 32, lm, g);
    av_step(sc, kk, lm, g, u, o);
    load_vfrag(u, vbase, kk + 64, lm, g);
    av_step(sc, kk + 32, lm, g, w2, o);
  }
  load_vfrag(w2, vbase, wv * 256 + 224, lm, g);
  av_step(sc, wv * 256 + 192, lm, g, u, o);
  av_step(sc, wv * 256 + 224, lm, g, w2, o);

#pragma unroll
  for (int j = 0; j < 4; ++j)
#pragma unroll
    for (int r = 0; r < 8; ++r)
      part[wv * 1024 + (g * 8 + r) * 64 + j * 16 + lm] = o[j][r];
  __syncthreads();

  for (int idx = tid; idx < 1024; idx += 256) {
    float s = 0.f;
#pragma unroll
    for (int w = 0; w < 8; ++w) s += part[w * 1024 + idx];
    const int row = idx >> 6, d = idx & 63;
    ctx[((size_t)(bb * SQn + q0 + row)) * Dn + hh * HDn + d] = s;
  }
}

// ---------------------------------------------------------------------------
// Residual + LayerNorm:  out = LN(qlin + oproj) * g + b.  One row per block.
// ---------------------------------------------------------------------------
__global__ __launch_bounds__(256, 1) void ln_kernel(
    const float* __restrict__ qlin, const float* __restrict__ op,
    const float* __restrict__ gam, const float* __restrict__ bet,
    float* __restrict__ out) {
  __shared__ float red[256];
  const int row = blockIdx.x, t = threadIdx.x;
  const float* a = qlin + (size_t)row * Dn;
  const float* b = op + (size_t)row * Dn;
  const float x0 = a[t] + b[t];
  const float x1 = a[t + 256] + b[t + 256];

  red[t] = x0 + x1;
  __syncthreads();
  for (int s = 128; s > 0; s >>= 1) {
    if (t < s) red[t] += red[t + s];
    __syncthreads();
  }
  const float mu = red[0] * (1.f / 512.f);
  __syncthreads();

  const float d0 = x0 - mu, d1 = x1 - mu;
  red[t] = d0 * d0 + d1 * d1;
  __syncthreads();
  for (int s = 128; s > 0; s >>= 1) {
    if (t < s) red[t] += red[t + s];
    __syncthreads();
  }
  const float rs = rsqrtf(red[0] * (1.f / 512.f) + 1e-5f);

  out[(size_t)row * Dn + t]       = d0 * rs * gam[t] + bet[t];
  out[(size_t)row * Dn + t + 256] = d1 * rs * gam[t + 256] + bet[t + 256];
}

// ---------------------------------------------------------------------------
extern "C" void kernel_launch(void* const* d_in, const int* in_sizes, int n_in,
                              void* d_out, int out_size, void* d_ws,
                              size_t ws_size, hipStream_t stream) {
  (void)in_sizes; (void)n_in; (void)out_size; (void)ws_size;

  const float* q  = (const float*)d_in[0];
  const float* k  = (const float*)d_in[1];
  const float* v  = (const float*)d_in[2];
  const float* Wq = (const float*)d_in[3];
  const float* bq = (const float*)d_in[4];
  const float* Wk = (const float*)d_in[5];
  const float* bk = (const float*)d_in[6];
  const float* Wv = (const float*)d_in[7];
  const float* bv = (const float*)d_in[8];
  const float* Wo = (const float*)d_in[9];
  const float* bo = (const float*)d_in[10];
  const float* lg = (const float*)d_in[11];
  const float* lb = (const float*)d_in[12];

  char* ws = (char*)d_ws;
  float*  qlin  = (float*)ws;                              // 16 MB
  __bf16* klin  = (__bf16*)(ws + (16u << 20));             //  8 MB
  __bf16* vT    = (__bf16*)(ws + (24u << 20));             //  8 MB
  float*  ctx   = (float*)(ws + (32u << 20));              // 16 MB
  float*  oproj = (float*)(ws + (48u << 20));              // 16 MB (total 64 MB)

  float* outp = (float*)d_out;
  float* attn = outp + (size_t)Bn * SQn * Dn;

  const dim3 gb(Dn / 64, (Bn * SQn) / 128);  // (8, 64)
  const int shmem_gemm = 64 * WSTR * 2;      // 67,584 B
  (void)hipFuncSetAttribute(reinterpret_cast<const void*>(proj_gemm_kernel),
                            hipFuncAttributeMaxDynamicSharedMemorySize,
                            shmem_gemm);

  proj_gemm_kernel<<<gb, 256, shmem_gemm, stream>>>(q, Wq, bq, qlin, nullptr,
                                                    nullptr);
  proj_gemm_kernel<<<gb, 256, shmem_gemm, stream>>>(k, Wk, bk, nullptr, klin,
                                                    nullptr);
  proj_gemm_kernel<<<gb, 256, shmem_gemm, stream>>>(v, Wv, bv, nullptr,
                                                    nullptr, vT);

  const int shmem_attn = 6400 + 16 * SSTR * 4 + 8 * 1024 * 4;  // 170,496 B
  (void)hipFuncSetAttribute(reinterpret_cast<const void*>(attn_kernel),
                            hipFuncAttributeMaxDynamicSharedMemorySize,
                            shmem_attn);
  attn_kernel<<<dim3(SQn / 16, Hn, Bn), 256, shmem_attn, stream>>>(
      qlin, klin, vT, ctx, attn);

  proj_gemm_kernel<<<gb, 256, shmem_gemm, stream>>>(ctx, Wo, bo, oproj,
                                                    nullptr, nullptr);
  ln_kernel<<<Bn * SQn, 256, 0, stream>>>(qlin, oproj, lg, lb, outp);
}
